// SimpleEnsembleNet_60078002536990
// MI455X (gfx1250) — matrix operand order repaired and checked by reading, not compiled, not run on hardware
//
#include <hip/hip_runtime.h>

typedef _Float16 half4  __attribute__((ext_vector_type(4)));
typedef _Float16 half8  __attribute__((ext_vector_type(8)));
typedef _Float16 half16 __attribute__((ext_vector_type(16)));
typedef float    f32x4  __attribute__((ext_vector_type(4)));
typedef float    f32x8  __attribute__((ext_vector_type(8)));

#define E_  8
#define D_  256
#define H_  256
#define O_  6
#define B_  32768
#define MT  64      // sample rows per block
#define TPB 64      // 2 waves; each wave owns 32 rows end-to-end (no barriers)

// ---------------------------------------------------------------- utilities

__device__ __forceinline__ float fast_tanh(float v) {
#if __has_builtin(__builtin_amdgcn_tanhf)
  return __builtin_amdgcn_tanhf(v);
#elif __has_builtin(__builtin_amdgcn_tanh_f32)
  return __builtin_amdgcn_tanh_f32(v);
#else
  return tanhf(v);
#endif
}

// ---------------------------------------------------------------- pass 1: bucketing

__global__ void init_counts_kernel(int* counts) {
  if (threadIdx.x < E_) counts[threadIdx.x] = 0;
}

__global__ void bucket_kernel(const float* __restrict__ noise,
                              int* __restrict__ counts,
                              unsigned* __restrict__ perm) {
  int i = blockIdx.x * blockDim.x + threadIdx.x;
  if (i >= B_) return;
  float g = noise[i];
  // ndtri(k/8), k = 1..7
  const float t0 = -1.1503494f, t1 = -0.6744898f, t2 = -0.3186394f, t3 = 0.0f,
              t4 =  0.3186394f, t5 =  0.6744898f, t6 =  1.1503494f;
  int idx = (g > t0) + (g > t1) + (g > t2) + (g > t3) + (g > t4) + (g > t5) + (g > t6);
  int pos = atomicAdd(&counts[idx], 1);
  perm[idx * B_ + pos] = (unsigned)i;
}

// ---------------------------------------------------------------- pass 2: weight repack f32 -> f16 B-layout
// tile layout: index(h, nT, kT, lane, e) = (((h*16 + nT)*8 + kT)*32 + lane)*16 + e
// maps element W[h][k][n] with n = nT*16 + (lane&15), k = kT*32 + (lane>>4)*8 + (e&7) + (e>>3)*16
// (matches V_WMMA_*_16X16X32 B-matrix VGPR layout: lanes 0-15 hold K 0..7,16..23; 16-31 hold K 8..15,24..31)

__global__ void convert_w_kernel(const float* __restrict__ W, _Float16* __restrict__ Wt) {
  int t = blockIdx.x * blockDim.x + threadIdx.x;   // 0 .. E_*65536
  int e    =  t        & 15;
  int lane = (t >> 4)  & 31;
  int kT   = (t >> 9)  & 7;
  int nT   = (t >> 12) & 15;
  int h    =  t >> 16;
  int n = nT * 16 + (lane & 15);
  int k = kT * 32 + (lane >> 4) * 8 + (e & 7) + (e >> 3) * 16;
  Wt[t] = (_Float16)W[(h * 256 + k) * 256 + n];
}

// ---------------------------------------------------------------- pass 3: per-head MLP via WMMA
// LDS swizzle: row of 256 halves = 32 chunks of 8 halves; chunk c of row r stored at c ^ (r & 31).
// A-frag (16-bit 16x32): lane -> row = rbase + (lane&15), K-base = kT*32 + (lane>>4)*8,
//   elements 0..7 = K..K+7, elements 8..15 = K+16..K+23  (two 16B LDS loads)
// Each wave processes TWO 16-row stripes so every B fragment feeds 2 WMMAs
// (halves L2 weight traffic, the dominant data stream).

__device__ __forceinline__ half16 load_afrag(const _Float16* __restrict__ bufIn,
                                             int row, int kb) {
  int swz = row & 31;
  int cl = ((kb) >> 3) ^ swz;
  int ch = ((kb + 16) >> 3) ^ swz;
  half8 alo = *(const half8*)&bufIn[row * 256 + cl * 8];
  half8 ahi = *(const half8*)&bufIn[row * 256 + ch * 8];
  return __builtin_shufflevector(alo, ahi,
           0, 1, 2, 3, 4, 5, 6, 7, 8, 9, 10, 11, 12, 13, 14, 15);
}

__device__ __forceinline__ void layer_gemm(const _Float16* __restrict__ bufIn,
                                           _Float16* __restrict__ bufOut,
                                           const _Float16* __restrict__ wt,
                                           const float* __restrict__ bias,
                                           int lane, int rbase) {
  const int row0 = rbase + (lane & 15);
  const int row1 = row0 + 16;
  const int hiK  = (lane >> 4) * 8;
  const half16* __restrict__ wv = (const half16*)wt;
#pragma unroll 1
  for (int nT = 0; nT < 16; ++nT) {
    float bcol = bias[nT * 16 + (lane & 15)];
    f32x8 acc0 = {bcol, bcol, bcol, bcol, bcol, bcol, bcol, bcol};
    f32x8 acc1 = acc0;
#pragma unroll
    for (int kT = 0; kT < 8; ++kT) {
      int kb = kT * 32 + hiK;
      half16 a0 = load_afrag(bufIn, row0, kb);
      half16 a1 = load_afrag(bufIn, row1, kb);
      half16 b  = wv[(nT * 8 + kT) * 32 + lane];
      acc0 = __builtin_amdgcn_wmma_f32_16x16x32_f16(
                 false, a0, false, b, (short)0, acc0, false, false);
      acc1 = __builtin_amdgcn_wmma_f32_16x16x32_f16(
                 false, a1, false, b, (short)0, acc1, false, false);
    }
    // C/D layout: lane&15 = column; lanes>=16 hold rows M+8; VGPR i = row offset
    int oR0 = rbase + ((lane >> 4) << 3);
    int col = nT * 16 + (lane & 15);
#pragma unroll
    for (int i = 0; i < 8; ++i) {
      int rr0   = oR0 + i;
      int phys0 = (col >> 3) ^ (rr0 & 31);
      bufOut[rr0 * 256 + phys0 * 8 + (col & 7)] = (_Float16)fast_tanh(acc0[i]);
      int rr1   = rr0 + 16;
      int phys1 = (col >> 3) ^ (rr1 & 31);
      bufOut[rr1 * 256 + phys1 * 8 + (col & 7)] = (_Float16)fast_tanh(acc1[i]);
    }
  }
}

__global__ void __launch_bounds__(TPB) mlp_kernel(
    const float* __restrict__ x,
    const unsigned* __restrict__ perm,
    const int* __restrict__ counts,
    const _Float16* __restrict__ W1t, const float* __restrict__ b1,
    const _Float16* __restrict__ W2t, const float* __restrict__ b2,
    const float* __restrict__ W3,     const float* __restrict__ b3,
    float* __restrict__ out)
{
  __shared__ _Float16 buf0[MT * 256];
  __shared__ _Float16 buf1[MT * 256];

  const int h     = blockIdx.y;
  const int cnt   = counts[h];
  const int tile0 = blockIdx.x * MT;
  if (tile0 >= cnt) return;             // uniform exit: WMMA below runs with EXEC all-ones

  const int lane  = threadIdx.x & 31;
  const int wave  = threadIdx.x >> 5;
  const int rbase = wave * 32;

  // -------- gather this wave's 32 x-rows into buf0 (f16, swizzled) --------
#pragma unroll 1
  for (int r32 = 0; r32 < 32; ++r32) {
    int row = rbase + r32;
    int g   = tile0 + row;
    int gi  = (g < cnt) ? g : (cnt - 1);         // clamp; padding rows never scattered
    unsigned s = perm[h * B_ + gi];
#pragma unroll
    for (int it = 0; it < 2; ++it) {
      int col = it * 128 + lane * 4;
      f32x4 v = *(const f32x4*)&x[(size_t)s * 256 + col];
      half4 hv = { (_Float16)v[0], (_Float16)v[1], (_Float16)v[2], (_Float16)v[3] };
      int phys = (col >> 3) ^ (row & 31);
      *(half4*)&buf0[row * 256 + phys * 8 + (col & 7)] = hv;
    }
  }

  // -------- layers 1 & 2: 256->256 f16 WMMA, tanh, f32 accumulate --------
  layer_gemm(buf0, buf1, W1t + (size_t)h * 65536, b1 + h * 256, lane, rbase);
  layer_gemm(buf1, buf0, W2t + (size_t)h * 65536, b2 + h * 256, lane, rbase);

  // -------- layer 3: 256 -> 6 (tiny; plain VALU), scatter per sample --------
  const float* __restrict__ w3h = W3 + (size_t)h * 256 * 6;
#pragma unroll 1
  for (int q = 0; q < 6; ++q) {
    int P   = q * 32 + lane;     // 192 (row, o) tasks over 32 lanes
    int r32 = P / 6;
    int o   = P % 6;
    int row = rbase + r32;
    int swz = row & 31;
    float acc = b3[h * 6 + o];
#pragma unroll 1
    for (int c = 0; c < 32; ++c) {
      int phys = c ^ swz;
      half8 hv = *(const half8*)&buf0[row * 256 + phys * 8];
#pragma unroll
      for (int j = 0; j < 8; ++j)
        acc += (float)hv[j] * w3h[(c * 8 + j) * 6 + o];
    }
    int g = tile0 + row;
    if (g < cnt) {
      unsigned s = perm[h * B_ + g];
      out[(size_t)s * 6 + o] = acc;
    }
  }
}

// ---------------------------------------------------------------- launch

extern "C" void kernel_launch(void* const* d_in, const int* in_sizes, int n_in,
                              void* d_out, int out_size, void* d_ws, size_t ws_size,
                              hipStream_t stream) {
  const float* x     = (const float*)d_in[0];
  const float* noise = (const float*)d_in[1];
  const float* W1    = (const float*)d_in[2];
  const float* b1    = (const float*)d_in[3];
  const float* W2    = (const float*)d_in[4];
  const float* b2    = (const float*)d_in[5];
  const float* W3    = (const float*)d_in[6];
  const float* b3    = (const float*)d_in[7];
  float* out = (float*)d_out;

  char* ws = (char*)d_ws;
  int*      counts = (int*)ws;                                   // 32 B (padded to 256)
  unsigned* perm   = (unsigned*)(ws + 256);                      // 8*32768*4 = 1 MB
  _Float16* W1t    = (_Float16*)(ws + 256 + (size_t)E_ * B_ * 4);// 1 MB
  _Float16* W2t    = W1t + (size_t)E_ * 256 * 256;               // 1 MB

  init_counts_kernel<<<1, 32, 0, stream>>>(counts);
  bucket_kernel<<<B_ / 256, 256, 0, stream>>>(noise, counts, perm);
  convert_w_kernel<<<(E_ * 256 * 256) / 256, 256, 0, stream>>>(W1, W1t);
  convert_w_kernel<<<(E_ * 256 * 256) / 256, 256, 0, stream>>>(W2, W2t);

  dim3 grid(B_ / MT, E_);
  mlp_kernel<<<grid, TPB, 0, stream>>>(x, perm, counts, W1t, b1, W2t, b2, W3, b3, out);
}
